// SCNCell_2147483648632
// MI455X (gfx1250) — compile-verified, matrix operand order checked by reading, not run on hardware
//
#include <hip/hip_runtime.h>
#include <hip/hip_bf16.h>

// ---------------------------------------------------------------------------
// SCN factorized-LSTM cell, MI455X (gfx1250), bf16 WMMA path.
// B=8192, IN=H=1024, F=512, 4F=2048.  ~206 GFLOP -> compute bound -> WMMA.
// Register-blocked 2x2 (stage A) / 2x1 (stage B) wave macro-tiles to get
// ~1 fragment load per v_wmma instead of ~3.
// ---------------------------------------------------------------------------

#define Bb   8192
#define INd  1024
#define Hh   1024
#define Ff   512
#define N4F  2048

typedef __attribute__((ext_vector_type(4)))  float  v4f;
typedef __attribute__((ext_vector_type(8)))  float  v8f;
typedef __attribute__((ext_vector_type(8)))  __bf16 v8bf;
typedef __attribute__((ext_vector_type(16))) __bf16 v16bf;

// Fragment loader for bf16 K-contiguous rows.
// ISA 7.12.2 16-bit A/B layout: lanes 0-15 hold K=[kb..kb+7] and [kb+16..kb+23]
// with kb=0; lanes 16-31 with kb=8. Caller pre-adds row*ld + kb.
__device__ __forceinline__ v16bf load_frag(const __bf16* __restrict__ p) {
    v8bf lo = *(const v8bf*)(p);        // K = kb .. kb+7
    v8bf hi = *(const v8bf*)(p + 16);   // K = kb+16 .. kb+23
    return __builtin_shufflevector(lo, hi,
        0, 1, 2, 3, 4, 5, 6, 7, 8, 9, 10, 11, 12, 13, 14, 15);
}

// Same fragment but converting f32 source data on the fly (fallback path).
__device__ __forceinline__ v16bf load_frag(const float* __restrict__ p) {
    v4f a0 = *(const v4f*)(p);
    v4f a1 = *(const v4f*)(p + 4);
    v4f b0 = *(const v4f*)(p + 16);
    v4f b1 = *(const v4f*)(p + 20);
    v16bf r;
#pragma unroll
    for (int i = 0; i < 4; ++i) {
        r[i]      = (__bf16)a0[i];
        r[4 + i]  = (__bf16)a1[i];
        r[8 + i]  = (__bf16)b0[i];
        r[12 + i] = (__bf16)b1[i];
    }
    return r;
}

__device__ __forceinline__ v8f wmma_bf16(v16bf a, v16bf b, v8f c) {
    return __builtin_amdgcn_wmma_f32_16x16x32_bf16(
        /*neg_a=*/false, a, /*neg_b=*/false, b,
        /*c_mod=*/(short)0, c, /*reuse_a=*/false, /*reuse_b=*/false);
}

// ---------------------------------------------------------------------------
// Prep kernel 1: f32 (K,N) -> bf16 (N,K) transpose+convert (LDS tiled).
// ---------------------------------------------------------------------------
__global__ void __launch_bounds__(256)
transpose_cvt_kernel(const float* __restrict__ in, __bf16* __restrict__ out,
                     int K, int N) {
    __shared__ float tile[32][33];
    const int n0 = blockIdx.x * 32;
    const int k0 = blockIdx.y * 32;
    const int tx = threadIdx.x & 31;
    const int ty = threadIdx.x >> 5;  // 0..7
#pragma unroll
    for (int r = 0; r < 4; ++r) {
        const int k = k0 + ty + 8 * r;
        tile[ty + 8 * r][tx] = in[(long)k * N + n0 + tx];  // coalesced in n
    }
    __syncthreads();
#pragma unroll
    for (int r = 0; r < 4; ++r) {
        const int n = n0 + ty + 8 * r;
        out[(long)n * K + k0 + tx] = (__bf16)tile[tx][ty + 8 * r];  // coalesced in k
    }
}

// ---------------------------------------------------------------------------
// Prep kernel 2: f32 -> bf16 same-layout convert.
// ---------------------------------------------------------------------------
__global__ void __launch_bounds__(256)
cvt_bf16_kernel(const float* __restrict__ in, __bf16* __restrict__ out, long n) {
    const long i = ((long)blockIdx.x * blockDim.x + threadIdx.x) * 4;
    if (i + 3 < n) {
        v4f v = *(const v4f*)(in + i);
#pragma unroll
        for (int t = 0; t < 4; ++t) out[i + t] = (__bf16)v[t];
    }
}

// ---------------------------------------------------------------------------
// Stage A: out[b,j] = bf16( (A1 @ B1T^T)[b,j] * (A2 @ B2T^T)[b,j] )
// A1,A2: (M,K) row-major (bf16 fast path or f32 fallback).
// B1T,B2T: bf16 (N,K) row-major (pre-transposed).
// Wave macro-tile 32x32: 2x2 subtiles x 2 matrices = 8 WMMAs per K-step from
// 8 fragment loads.
// ---------------------------------------------------------------------------
template <typename AT>
__global__ void __launch_bounds__(256)
fgemm_mul_kernel(const AT* __restrict__ A1, const AT* __restrict__ A2,
                 const __bf16* __restrict__ B1T, const __bf16* __restrict__ B2T,
                 __bf16* __restrict__ out, int K, int N) {
    const int lane = threadIdx.x & 31;
    const int wave = threadIdx.x >> 5;
    const long tile = (long)blockIdx.x * 8 + wave;
    const int tilesN = N >> 5;                 // 32-wide macro tiles
    const int m0 = (int)(tile / tilesN) * 32;
    const int n0 = (int)(tile % tilesN) * 32;
    const int r  = lane & 15;
    const int kb = (lane >> 4) << 3;           // 0 or 8

    const AT*     a1p[2];
    const AT*     a2p[2];
    const __bf16* b1p[2];
    const __bf16* b2p[2];
#pragma unroll
    for (int i = 0; i < 2; ++i) {
        a1p[i] = A1  + (long)(m0 + 16 * i + r) * K + kb;
        a2p[i] = A2  + (long)(m0 + 16 * i + r) * K + kb;
        b1p[i] = B1T + (long)(n0 + 16 * i + r) * K + kb;
        b2p[i] = B2T + (long)(n0 + 16 * i + r) * K + kb;
    }

    v8f acc1[2][2], acc2[2][2];
#pragma unroll
    for (int i = 0; i < 2; ++i)
#pragma unroll
        for (int j = 0; j < 2; ++j)
#pragma unroll
            for (int v = 0; v < 8; ++v) { acc1[i][j][v] = 0.f; acc2[i][j][v] = 0.f; }

#pragma unroll 2
    for (int k0 = 0; k0 < K; k0 += 32) {
        __builtin_prefetch(a1p[0] + k0 + 256, 0, 3);  // global_prefetch_b8 (near)
        __builtin_prefetch(a2p[0] + k0 + 256, 0, 3);
        __builtin_prefetch(b1p[0] + k0 + 256, 0, 3);
        __builtin_prefetch(b2p[0] + k0 + 256, 0, 3);
        v16bf a1f[2], a2f[2], b1f[2], b2f[2];
#pragma unroll
        for (int i = 0; i < 2; ++i) {
            a1f[i] = load_frag(a1p[i] + k0);
            a2f[i] = load_frag(a2p[i] + k0);
            b1f[i] = load_frag(b1p[i] + k0);
            b2f[i] = load_frag(b2p[i] + k0);
        }
#pragma unroll
        for (int i = 0; i < 2; ++i)
#pragma unroll
            for (int j = 0; j < 2; ++j) {
                acc1[i][j] = wmma_bf16(a1f[i], b1f[j], acc1[i][j]);
                acc2[i][j] = wmma_bf16(a2f[i], b2f[j], acc2[i][j]);
            }
    }

    // C/D layout: VGPR v -> row sub_m0 + v + 8*(lane>=16), col sub_n0 + (lane&15).
    const int rowh = (lane >> 4) << 3;
#pragma unroll
    for (int i = 0; i < 2; ++i)
#pragma unroll
        for (int j = 0; j < 2; ++j) {
            const int col = n0 + 16 * j + r;
#pragma unroll
            for (int v = 0; v < 8; ++v) {
                const int row = m0 + 16 * i + rowh + v;
                out[(long)row * N + col] = (__bf16)(acc1[i][j][v] * acc2[i][j][v]);
            }
        }
}

// ---------------------------------------------------------------------------
// Stage B: per 32x16 (b,h) macro-tile, accumulate all 4 gates from
// U@Wic^T + R@Whc^T (K = F = 512 each source), then fused LSTM gating.
// 2x1 row blocking: each weight fragment feeds 2 WMMAs.
// out: [h (B*H) | c (B*H)] f32.
// ---------------------------------------------------------------------------
__global__ void __launch_bounds__(256)
gate_kernel(const __bf16* __restrict__ U,   const __bf16* __restrict__ R,
            const __bf16* __restrict__ Wic, const __bf16* __restrict__ Whc,
            const float* __restrict__ bias_ih, const float* __restrict__ bias_hh,
            const float* __restrict__ c0, float* __restrict__ out) {
    const int lane = threadIdx.x & 31;
    const int wave = threadIdx.x >> 5;
    const long tile = (long)blockIdx.x * 8 + wave;
    const int tilesN = Hh >> 4;                // 64 column tiles
    const int m0 = (int)(tile / tilesN) * 32;  // 32-row macro tile
    const int n0 = (int)(tile % tilesN) * 16;
    const int r  = lane & 15;
    const int kb = (lane >> 4) << 3;

    const __bf16* up[2];
    const __bf16* rp[2];
#pragma unroll
    for (int i = 0; i < 2; ++i) {
        up[i] = U + (long)(m0 + 16 * i + r) * N4F + kb;
        rp[i] = R + (long)(m0 + 16 * i + r) * N4F + kb;
    }
    const __bf16* icp = Wic + (long)(n0 + r) * N4F + kb;
    const __bf16* hcp = Whc + (long)(n0 + r) * N4F + kb;

    v8f acc[4][2];
#pragma unroll
    for (int g = 0; g < 4; ++g)
#pragma unroll
        for (int i = 0; i < 2; ++i)
#pragma unroll
            for (int v = 0; v < 8; ++v) acc[g][i][v] = 0.f;

#pragma unroll
    for (int g = 0; g < 4; ++g) {
        const int goff = g * Ff;
#pragma unroll 2
        for (int k0 = 0; k0 < Ff; k0 += 32) {   // input-side contribution
            __builtin_prefetch(up[0] + goff + k0 + 256, 0, 3);
            v16bf b = load_frag(icp + goff + k0);
#pragma unroll
            for (int i = 0; i < 2; ++i) {
                v16bf a = load_frag(up[i] + goff + k0);
                acc[g][i] = wmma_bf16(a, b, acc[g][i]);
            }
        }
#pragma unroll 2
        for (int k0 = 0; k0 < Ff; k0 += 32) {   // recurrent-side contribution
            __builtin_prefetch(rp[0] + goff + k0 + 256, 0, 3);
            v16bf b = load_frag(hcp + goff + k0);
#pragma unroll
            for (int i = 0; i < 2; ++i) {
                v16bf a = load_frag(rp[i] + goff + k0);
                acc[g][i] = wmma_bf16(a, b, acc[g][i]);
            }
        }
    }

    const int col = n0 + r;
    float bsum[4];
#pragma unroll
    for (int g = 0; g < 4; ++g)
        bsum[g] = bias_ih[g * Hh + col] + bias_hh[g * Hh + col];

    const int rowh = (lane >> 4) << 3;
#pragma unroll
    for (int i = 0; i < 2; ++i) {
#pragma unroll
        for (int v = 0; v < 8; ++v) {
            const int row = m0 + 16 * i + rowh + v;
            const float pi = acc[0][i][v] + bsum[0];
            const float pf = acc[1][i][v] + bsum[1];
            const float po = acc[2][i][v] + bsum[2];
            const float pc = acc[3][i][v] + bsum[3];
            const float ig = 1.f / (1.f + __expf(-pi));
            const float fg = 1.f / (1.f + __expf(-pf));
            const float og = 1.f / (1.f + __expf(-po));
            const float ct = tanhf(pc);
            const float cp = c0[(long)row * Hh + col];
            const float c  = fg * cp + ig * ct;
            const float h  = og * tanhf(c);
            out[(long)row * Hh + col]                 = h;
            out[(long)Bb * Hh + (long)row * Hh + col] = c;
        }
    }
}

// ---------------------------------------------------------------------------
extern "C" void kernel_launch(void* const* d_in, const int* in_sizes, int n_in,
                              void* d_out, int out_size, void* d_ws, size_t ws_size,
                              hipStream_t stream) {
    const float* wemb = (const float*)d_in[0];
    const float* tag  = (const float*)d_in[1];
    const float* h0   = (const float*)d_in[2];
    const float* c0   = (const float*)d_in[3];
    const float* Wia  = (const float*)d_in[4];
    const float* Wib  = (const float*)d_in[5];
    const float* Wic  = (const float*)d_in[6];
    const float* Wha  = (const float*)d_in[7];
    const float* Whb  = (const float*)d_in[8];
    const float* Whc  = (const float*)d_in[9];
    const float* bih  = (const float*)d_in[10];
    const float* bhh  = (const float*)d_in[11];

    // Workspace carve-up (bf16 buffers).
    char* ws = (char*)d_ws;
    size_t o = 0;
    const size_t wTsz = (size_t)N4F * INd * sizeof(__bf16);   // 4 MB each
    __bf16* WiaT = (__bf16*)(ws + o); o += wTsz;
    __bf16* WibT = (__bf16*)(ws + o); o += wTsz;
    __bf16* WhaT = (__bf16*)(ws + o); o += wTsz;
    __bf16* WhbT = (__bf16*)(ws + o); o += wTsz;
    __bf16* WicB = (__bf16*)(ws + o); o += wTsz;
    __bf16* WhcB = (__bf16*)(ws + o); o += wTsz;
    const size_t usz = (size_t)Bb * N4F * sizeof(__bf16);     // 33.5 MB each
    __bf16* Ubuf = (__bf16*)(ws + o); o += usz;
    __bf16* Rbuf = (__bf16*)(ws + o); o += usz;

    // Optional activation pre-convert (needs +3 x 16.8 MB workspace).
    const size_t actsz = (size_t)Bb * INd * sizeof(__bf16);
    const bool pre_cvt = (ws_size >= o + 3 * actsz);
    __bf16* Xbf = nullptr; __bf16* Tbf = nullptr; __bf16* Hbf = nullptr;
    if (pre_cvt) {
        Xbf = (__bf16*)(ws + o); o += actsz;
        Tbf = (__bf16*)(ws + o); o += actsz;
        Hbf = (__bf16*)(ws + o); o += actsz;
    }

    // Prep: transpose+cvt factor weights; cvt contraction weights.
    dim3 tgrid(N4F / 32, INd / 32);
    transpose_cvt_kernel<<<tgrid, 256, 0, stream>>>(Wia, WiaT, INd, N4F);
    transpose_cvt_kernel<<<tgrid, 256, 0, stream>>>(Wib, WibT, Hh, N4F);
    transpose_cvt_kernel<<<tgrid, 256, 0, stream>>>(Wha, WhaT, Hh, N4F);
    transpose_cvt_kernel<<<tgrid, 256, 0, stream>>>(Whb, WhbT, Hh, N4F);
    const long wEl = (long)Hh * N4F;
    const unsigned cblk = (unsigned)((wEl / 4 + 255) / 256);
    cvt_bf16_kernel<<<cblk, 256, 0, stream>>>(Wic, WicB, wEl);
    cvt_bf16_kernel<<<cblk, 256, 0, stream>>>(Whc, WhcB, wEl);

    const unsigned ablocks = (Bb / 32) * (N4F / 32) / 8;  // 2048
    if (pre_cvt) {
        const long aEl = (long)Bb * INd;
        const unsigned ablk = (unsigned)((aEl / 4 + 255) / 256);
        cvt_bf16_kernel<<<ablk, 256, 0, stream>>>(wemb, Xbf, aEl);
        cvt_bf16_kernel<<<ablk, 256, 0, stream>>>(tag,  Tbf, aEl);
        cvt_bf16_kernel<<<ablk, 256, 0, stream>>>(h0,   Hbf, aEl);
        fgemm_mul_kernel<__bf16><<<ablocks, 256, 0, stream>>>(
            Xbf, Tbf, WiaT, WibT, Ubuf, INd, N4F);
        fgemm_mul_kernel<__bf16><<<ablocks, 256, 0, stream>>>(
            Hbf, Tbf, WhaT, WhbT, Rbuf, Hh, N4F);
    } else {
        fgemm_mul_kernel<float><<<ablocks, 256, 0, stream>>>(
            wemb, tag, WiaT, WibT, Ubuf, INd, N4F);
        fgemm_mul_kernel<float><<<ablocks, 256, 0, stream>>>(
            h0, tag, WhaT, WhbT, Rbuf, Hh, N4F);
    }

    // Stage B: fused 4-gate contraction + LSTM gating -> h, c
    const unsigned gblocks = (Bb / 32) * (Hh / 16) / 8;   // 2048
    gate_kernel<<<gblocks, 256, 0, stream>>>(Ubuf, Rbuf, WicB, WhcB, bih, bhh, c0,
                                             (float*)d_out);
}
